// SparseAttentionBlock_20203526160831
// MI455X (gfx1250) — matrix operand order repaired and checked by reading, not compiled, not run on hardware
//
#include <hip/hip_runtime.h>
#include <stdint.h>
#include <stddef.h>

#define B_  8
#define T_  1024
#define D_  256
#define H_  8
#define KD_ 256
#define HK_ 2048
#define FF_ 1024
#define BT_ (B_*T_)

typedef __attribute__((ext_vector_type(16))) __bf16 bf16x16;
typedef __attribute__((ext_vector_type(8)))  float  f32x8;

union FragBF {
    unsigned int   u[8];
    unsigned short us[16];
    bf16x16        v;
};

__device__ __forceinline__ unsigned short f2bf(float f) {
    unsigned int u = __float_as_uint(f);
    unsigned int r = u + 0x7FFFu + ((u >> 16) & 1u);   // RNE
    return (unsigned short)(r >> 16);
}
__device__ __forceinline__ float bf2f(unsigned short s) {
    return __uint_as_float(((unsigned int)s) << 16);
}
__device__ __forceinline__ float wredsum32(float x) {
    for (int m = 1; m < 32; m <<= 1) x += __shfl_xor(x, m, 32);
    return x;
}
__device__ __forceinline__ float wredmax16(float x) {
    for (int m = 1; m < 16; m <<= 1) x = fmaxf(x, __shfl_xor(x, m, 32));
    return x;
}
__device__ __forceinline__ float wredsum16(float x) {
    for (int m = 1; m < 16; m <<= 1) x += __shfl_xor(x, m, 32);
    return x;
}

// ---------------------------------------------------------------------------
// Generic tiled GEMM: C[M,N] = act(A[M,K] @ B[K,N] + bias (+ C if ACC))
// A: f32 or bf16 (ushort). B: f32 weights. 128x128 tile, BK=32, 8 waves.
// ---------------------------------------------------------------------------
template<bool A_BF16, bool OUT_BF16, bool RELU, bool ACC>
__global__ __launch_bounds__(256) void gemm_wmma(
    const void* __restrict__ Ap, const float* __restrict__ Bp,
    const float* __restrict__ bias, void* __restrict__ Cp,
    int M, int N, int K)
{
    __shared__ unsigned short As[128][36];   // stride 36 (even -> 4B-aligned pairs)
    __shared__ unsigned short Bs[32][132];

    const int tid  = threadIdx.x;
    const int lane = tid & 31;
    const int wid  = tid >> 5;
    const int wm   = (wid >> 2) * 64;        // wave M offset within tile
    const int wn   = (wid & 3)  * 32;        // wave N offset within tile
    const int m0   = blockIdx.y * 128;
    const int n0   = blockIdx.x * 128;
    const int mL   = lane & 15;              // row / col within fragment
    const int hL   = lane >> 4;              // half-wave index

    f32x8 acc[4][2];
    #pragma unroll
    for (int r = 0; r < 4; ++r)
        #pragma unroll
        for (int c = 0; c < 2; ++c)
            acc[r][c] = (f32x8){0.f,0.f,0.f,0.f,0.f,0.f,0.f,0.f};

    const float*          Af = (const float*)Ap;
    const unsigned short* Ab = (const unsigned short*)Ap;

    for (int k0 = 0; k0 < K; k0 += 32) {
        // Stage A tile (128x32) as bf16
        #pragma unroll
        for (int i = 0; i < 16; ++i) {
            int idx = i * 256 + tid;
            int r = idx >> 5, c = idx & 31;
            int row = m0 + r;
            unsigned short v = 0;
            if (row < M) {
                size_t off = (size_t)row * K + (size_t)(k0 + c);
                v = A_BF16 ? Ab[off] : f2bf(Af[off]);
            }
            As[r][c] = v;
        }
        // Stage B tile (32x128) as bf16
        #pragma unroll
        for (int i = 0; i < 16; ++i) {
            int idx = i * 256 + tid;
            int r = idx >> 7, c = idx & 127;
            Bs[r][c] = f2bf(Bp[(size_t)(k0 + r) * N + (size_t)(n0 + c)]);
        }
        __syncthreads();

        FragBF afr[4], bfr[2];
        #pragma unroll
        for (int r = 0; r < 4; ++r) {
            #pragma unroll
            for (int p = 0; p < 8; ++p) {
                int kk = hL * 16 + p * 2;
                afr[r].u[p] = *(const unsigned int*)&As[wm + r * 16 + mL][kk];
            }
        }
        #pragma unroll
        for (int c = 0; c < 2; ++c) {
            #pragma unroll
            for (int j = 0; j < 8; ++j) {
                int kk = hL * 16 + 2 * j;
                bfr[c].us[2 * j]     = Bs[kk][wn + c * 16 + mL];
                bfr[c].us[2 * j + 1] = Bs[kk + 1][wn + c * 16 + mL];
            }
        }
        #pragma unroll
        for (int r = 0; r < 4; ++r)
            #pragma unroll
            for (int c = 0; c < 2; ++c)
                acc[r][c] = __builtin_amdgcn_wmma_f32_16x16x32_bf16(
                    false, afr[r].v, false, bfr[c].v, (short)0, acc[r][c], false, false);
        __syncthreads();
    }

    float*          Cf = (float*)Cp;
    unsigned short* Cb = (unsigned short*)Cp;
    #pragma unroll
    for (int r4 = 0; r4 < 4; ++r4) {
        #pragma unroll
        for (int c2 = 0; c2 < 2; ++c2) {
            #pragma unroll
            for (int r = 0; r < 8; ++r) {
                int row = m0 + wm + r4 * 16 + r + 8 * hL;
                int col = n0 + wn + c2 * 16 + mL;
                if (row < M) {
                    float v = acc[r4][c2][r];
                    if (bias) v += bias[col];
                    size_t off = (size_t)row * N + col;
                    if (ACC)  v += Cf[off];
                    if (RELU) v = fmaxf(v, 0.0f);
                    if (OUT_BF16) Cb[off] = f2bf(v);
                    else          Cf[off] = v;
                }
            }
        }
    }
}

// ---------------------------------------------------------------------------
// Local windowed causal attention (WIN=64), one wave per 16-query block.
// Q/K/V bf16 [BT, HK], output O bf16 [BT, HK]. Scale = 1/sqrt(256) = 1/16.
// Keys covered by 3 blocks of 32 starting at t0-64 (mask handles the rest).
// ---------------------------------------------------------------------------
__global__ __launch_bounds__(128) void local_attn_kernel(
    const unsigned short* __restrict__ Q, const unsigned short* __restrict__ Kt,
    const unsigned short* __restrict__ V, unsigned short* __restrict__ O)
{
    __shared__ unsigned short ps[4][16][96];   // per-wave P tile (16 x 96) bf16

    const int lane = threadIdx.x & 31;
    const int wid  = threadIdx.x >> 5;
    const int qb   = blockIdx.x * 4 + wid;     // 0 .. 4095
    const int tblk = qb & 63;
    const int bh   = qb >> 6;
    const int h    = bh & 7;
    const int b    = bh >> 3;
    const int t0   = tblk * 16;
    const int mL   = lane & 15;
    const int hL   = lane >> 4;
    const size_t rowBase = (size_t)b * T_;
    const size_t cbase   = (size_t)h * KD_;

    // Preload Q fragments along K (8 x 16x32)
    FragBF qf[8];
    {
        size_t qrow = (rowBase + t0 + mL) * HK_ + cbase;
        #pragma unroll
        for (int f = 0; f < 8; ++f) {
            #pragma unroll
            for (int p = 0; p < 8; ++p) {
                int kk = f * 32 + hL * 16 + p * 2;
                qf[f].u[p] = *(const unsigned int*)&Q[qrow + kk];
            }
        }
    }

    // Scores S = Q K^T for 3 key blocks of 32 (each = two 16x16 WMMA tiles)
    f32x8 sc[3][2];
    #pragma unroll
    for (int kb = 0; kb < 3; ++kb) {
        int s0 = t0 - 64 + kb * 32;
        f32x8 c0 = (f32x8){0.f,0.f,0.f,0.f,0.f,0.f,0.f,0.f};
        f32x8 c1 = c0;
        int sr0 = s0 + mL;      if (sr0 < 0) sr0 = 0;
        int sr1 = s0 + 16 + mL; if (sr1 < 0) sr1 = 0;
        size_t kr0 = (rowBase + sr0) * HK_ + cbase;
        size_t kr1 = (rowBase + sr1) * HK_ + cbase;
        #pragma unroll
        for (int f = 0; f < 8; ++f) {
            FragBF b0, b1;
            #pragma unroll
            for (int j = 0; j < 8; ++j) {
                int kk = f * 32 + hL * 16 + 2 * j;
                b0.u[j] = *(const unsigned int*)&Kt[kr0 + kk];
                b1.u[j] = *(const unsigned int*)&Kt[kr1 + kk];
            }
            c0 = __builtin_amdgcn_wmma_f32_16x16x32_bf16(false, qf[f].v, false, b0.v, (short)0, c0, false, false);
            c1 = __builtin_amdgcn_wmma_f32_16x16x32_bf16(false, qf[f].v, false, b1.v, (short)0, c1, false, false);
        }
        sc[kb][0] = c0;
        sc[kb][1] = c1;
    }

    // Scale, mask, softmax (row stats reduced over the 16 lanes sharing a row)
    float rmax[8];
    #pragma unroll
    for (int r = 0; r < 8; ++r) rmax[r] = -3.0e38f;
    #pragma unroll
    for (int kb = 0; kb < 3; ++kb) {
        int s0 = t0 - 64 + kb * 32;
        #pragma unroll
        for (int half = 0; half < 2; ++half) {
            #pragma unroll
            for (int r = 0; r < 8; ++r) {
                int i = t0 + r + 8 * hL;
                int j = s0 + half * 16 + mL;
                float v = sc[kb][half][r] * 0.0625f;
                bool ok = (j >= 0) && (j <= i) && (j > i - 64);
                v = ok ? v : -1.0e9f;
                sc[kb][half][r] = v;
                rmax[r] = fmaxf(rmax[r], v);
            }
        }
    }
    #pragma unroll
    for (int r = 0; r < 8; ++r) rmax[r] = wredmax16(rmax[r]);

    float rsum[8];
    #pragma unroll
    for (int r = 0; r < 8; ++r) rsum[r] = 0.f;
    #pragma unroll
    for (int kb = 0; kb < 3; ++kb) {
        #pragma unroll
        for (int half = 0; half < 2; ++half) {
            #pragma unroll
            for (int r = 0; r < 8; ++r) {
                float e = __expf(sc[kb][half][r] - rmax[r]);
                rsum[r] += e;
                ps[wid][r + 8 * hL][kb * 32 + half * 16 + mL] = f2bf(e);
            }
        }
    }
    #pragma unroll
    for (int r = 0; r < 8; ++r) rsum[r] = wredsum16(rsum[r]);
    __syncthreads();

    // O = P V, one 16x16 accumulator per 16-wide KD chunk
    for (int n0 = 0; n0 < KD_; n0 += 16) {
        f32x8 o = (f32x8){0.f,0.f,0.f,0.f,0.f,0.f,0.f,0.f};
        #pragma unroll
        for (int kb = 0; kb < 3; ++kb) {
            int s0 = t0 - 64 + kb * 32;
            FragBF pa, bv;
            #pragma unroll
            for (int p = 0; p < 8; ++p) {
                int kk = hL * 16 + p * 2;
                pa.u[p] = *(const unsigned int*)&ps[wid][mL][kb * 32 + kk];
            }
            #pragma unroll
            for (int j = 0; j < 8; ++j) {
                int kk = hL * 16 + 2 * j;
                int sa = s0 + kk;     if (sa < 0) sa = 0;
                int sb = s0 + kk + 1; if (sb < 0) sb = 0;
                bv.us[2 * j]     = V[(rowBase + sa) * HK_ + cbase + n0 + mL];
                bv.us[2 * j + 1] = V[(rowBase + sb) * HK_ + cbase + n0 + mL];
            }
            o = __builtin_amdgcn_wmma_f32_16x16x32_bf16(false, pa.v, false, bv.v, (short)0, o, false, false);
        }
        #pragma unroll
        for (int r = 0; r < 8; ++r) {
            int trow = t0 + r + 8 * hL;
            O[(rowBase + trow) * HK_ + cbase + n0 + mL] = f2bf(o[r] / rsum[r]);
        }
    }
}

// ---------------------------------------------------------------------------
// g2 attention: 2 global queries attend over all T keys. One block per
// (b, qg, h). gQ f32 [16,HK], K/V bf16 [BT,HK] -> oG bf16 [16,HK].
// ---------------------------------------------------------------------------
__global__ __launch_bounds__(256) void g2_attn_kernel(
    const float* __restrict__ gQ, const unsigned short* __restrict__ Kt,
    const unsigned short* __restrict__ V, unsigned short* __restrict__ oG)
{
    __shared__ float qs[256];
    __shared__ float aw[1024];
    __shared__ float red[256];

    const int tid = threadIdx.x;
    const int b  = blockIdx.x >> 4;
    const int qg = (blockIdx.x >> 3) & 1;
    const int h  = blockIdx.x & 7;

    qs[tid] = gQ[(size_t)(b * 2 + qg) * HK_ + h * KD_ + tid];
    __syncthreads();

    float sl[4];
    #pragma unroll
    for (int j2 = 0; j2 < 4; ++j2) {
        int t = tid + 256 * j2;
        const unsigned short* kr = Kt + (size_t)(b * T_ + t) * HK_ + h * KD_;
        float d = 0.f;
        for (int k = 0; k < 256; ++k) d += qs[k] * bf2f(kr[k]);
        sl[j2] = d * 0.0625f;
    }
    float lm = fmaxf(fmaxf(sl[0], sl[1]), fmaxf(sl[2], sl[3]));
    red[tid] = lm; __syncthreads();
    for (int s = 128; s > 0; s >>= 1) { if (tid < s) red[tid] = fmaxf(red[tid], red[tid + s]); __syncthreads(); }
    float gm = red[0];
    __syncthreads();

    float ls = 0.f;
    #pragma unroll
    for (int j2 = 0; j2 < 4; ++j2) {
        float e = __expf(sl[j2] - gm);
        aw[tid + 256 * j2] = e;
        ls += e;
    }
    red[tid] = ls; __syncthreads();
    for (int s = 128; s > 0; s >>= 1) { if (tid < s) red[tid] += red[tid + s]; __syncthreads(); }
    float inv = 1.0f / red[0];

    float acc = 0.f;
    for (int t = 0; t < 1024; ++t)
        acc += aw[t] * bf2f(V[(size_t)(b * T_ + t) * HK_ + h * KD_ + tid]);
    oG[(size_t)(b * 2 + qg) * HK_ + h * KD_ + tid] = f2bf(acc * inv);
}

// ---------------------------------------------------------------------------
// gx attention: every token attends over the 2 global key/values.
// One wave per (row, h). Q2 bf16 [BT,HK], K2/V2 f32 [16,HK] -> o2 bf16.
// ---------------------------------------------------------------------------
__global__ __launch_bounds__(256) void gx_attn_kernel(
    const unsigned short* __restrict__ Q2, const float* __restrict__ K2,
    const float* __restrict__ V2, unsigned short* __restrict__ o2)
{
    const int lane = threadIdx.x & 31;
    const int wid  = threadIdx.x >> 5;
    const int idx  = blockIdx.x * 8 + wid;    // over BT*H
    const int h    = idx & 7;
    const int row  = idx >> 3;
    const int b    = row >> 10;

    const unsigned short* qr = Q2 + (size_t)row * HK_ + h * KD_ + lane * 8;
    const float* k0p = K2 + (size_t)(b * 2 + 0) * HK_ + h * KD_ + lane * 8;
    const float* k1p = K2 + (size_t)(b * 2 + 1) * HK_ + h * KD_ + lane * 8;
    const float* v0p = V2 + (size_t)(b * 2 + 0) * HK_ + h * KD_ + lane * 8;
    const float* v1p = V2 + (size_t)(b * 2 + 1) * HK_ + h * KD_ + lane * 8;

    float q[8];
    float s0 = 0.f, s1 = 0.f;
    #pragma unroll
    for (int i = 0; i < 8; ++i) {
        q[i] = bf2f(qr[i]);
        s0 += q[i] * k0p[i];
        s1 += q[i] * k1p[i];
    }
    s0 = wredsum32(s0) * 0.0625f;
    s1 = wredsum32(s1) * 0.0625f;
    float m  = fmaxf(s0, s1);
    float e0 = __expf(s0 - m), e1 = __expf(s1 - m);
    float inv = 1.0f / (e0 + e1);
    float a0 = e0 * inv, a1 = e1 * inv;

    unsigned short* op = o2 + (size_t)row * HK_ + h * KD_ + lane * 8;
    #pragma unroll
    for (int i = 0; i < 8; ++i) op[i] = f2bf(a0 * v0p[i] + a1 * v1p[i]);
}

// ---------------------------------------------------------------------------
// out = LayerNorm(a + b) * gamma + beta. One wave per row (D=256, 8/lane).
// ---------------------------------------------------------------------------
__global__ __launch_bounds__(256) void ln_add_kernel(
    const float* __restrict__ a, const float* __restrict__ bsrc,
    const float* __restrict__ gamma, const float* __restrict__ beta,
    float* __restrict__ out, int rows)
{
    const int lane = threadIdx.x & 31;
    const int wid  = threadIdx.x >> 5;
    const int row  = blockIdx.x * 8 + wid;
    if (row >= rows) return;
    const size_t base = (size_t)row * D_ + lane * 8;

    float v[8], s = 0.f;
    #pragma unroll
    for (int i = 0; i < 8; ++i) { v[i] = a[base + i] + bsrc[base + i]; s += v[i]; }
    s = wredsum32(s);
    float mean = s * (1.0f / 256.0f);
    float s2 = 0.f;
    #pragma unroll
    for (int i = 0; i < 8; ++i) { float d = v[i] - mean; s2 += d * d; }
    s2 = wredsum32(s2);
    float invs = rsqrtf(s2 * (1.0f / 256.0f) + 1e-6f);
    #pragma unroll
    for (int i = 0; i < 8; ++i) {
        int c = lane * 8 + i;
        out[base + i] = (v[i] - mean) * invs * gamma[c] + beta[c];
    }
}

// Gather g = x[:, T-2:, :]  -> [16, 256]
__global__ void gather_g_kernel(const float* __restrict__ x, float* __restrict__ g)
{
    int b = blockIdx.x >> 1, qg = blockIdx.x & 1;
    g[(size_t)blockIdx.x * D_ + threadIdx.x] =
        x[((size_t)(b * T_ + (T_ - 2) + qg)) * D_ + threadIdx.x];
}

// ---------------------------------------------------------------------------
extern "C" void kernel_launch(void* const* d_in, const int* in_sizes, int n_in,
                              void* d_out, int out_size, void* d_ws, size_t ws_size,
                              hipStream_t stream)
{
    const float* x    = (const float*)d_in[0];
    const float* lWq  = (const float*)d_in[1];
    const float* lbq  = (const float*)d_in[2];
    const float* lWk  = (const float*)d_in[3];
    const float* lbk  = (const float*)d_in[4];
    const float* lWv  = (const float*)d_in[5];
    const float* lbv  = (const float*)d_in[6];
    const float* lWo  = (const float*)d_in[7];   // [2048, 256] flattened
    const float* lbo  = (const float*)d_in[8];
    const float* gWq  = (const float*)d_in[9];
    const float* gbq  = (const float*)d_in[10];
    const float* gWk  = (const float*)d_in[11];
    const float* gbk  = (const float*)d_in[12];
    const float* gWv  = (const float*)d_in[13];
    const float* gbv  = (const float*)d_in[14];
    const float* gWo  = (const float*)d_in[15];
    const float* gbo  = (const float*)d_in[16];
    const float* fW1  = (const float*)d_in[17];
    const float* fb1  = (const float*)d_in[18];
    const float* fW2  = (const float*)d_in[19];
    const float* fb2  = (const float*)d_in[20];
    const float* ln1g = (const float*)d_in[21];
    const float* ln1b = (const float*)d_in[22];
    const float* ln2g = (const float*)d_in[23];
    const float* ln2b = (const float*)d_in[24];

    char* ws = (char*)d_ws;
    const size_t MB = (size_t)1 << 20;
    unsigned short* lQ   = (unsigned short*)(ws + 0 * MB);     // 32MB
    unsigned short* lK   = (unsigned short*)(ws + 32 * MB);    // 32MB
    unsigned short* lV   = (unsigned short*)(ws + 64 * MB);    // 32MB
    unsigned short* lO   = (unsigned short*)(ws + 96 * MB);    // 32MB
    float*          attn = (float*)(ws + 128 * MB);            // 8MB
    float*          out1 = (float*)(ws + 136 * MB);            // 8MB
    float*          ffn  = (float*)(ws + 144 * MB);            // 8MB
    char* sm = ws + 152 * MB;
    float*          gbuf = (float*)sm;           sm += 16 * 256 * 4;
    float*          gQ   = (float*)sm;           sm += 16 * 2048 * 4;
    unsigned short* oG   = (unsigned short*)sm;  sm += 16 * 2048 * 2;
    float*          g2   = (float*)sm;           sm += 16 * 256 * 4;
    float*          K2   = (float*)sm;           sm += 16 * 2048 * 4;
    float*          V2   = (float*)sm;           sm += 16 * 2048 * 4;
    // Region reuse (earlier consumers finished before these are written):
    unsigned short* gK = lQ;    // phase G
    unsigned short* gV = lK;
    unsigned short* Q2 = lV;
    unsigned short* o2 = lO;
    unsigned short* h1 = (unsigned short*)(ws + 0 * MB);  // FFN hidden, 16MB

    dim3 blk(256);
    auto gg = [](int M, int N) { return dim3((unsigned)(N / 128), (unsigned)((M + 127) / 128)); };

    // ---- Local attention branch ----
    gemm_wmma<false, true, false, false><<<gg(BT_, HK_), blk, 0, stream>>>(x, lWq, lbq, lQ, BT_, HK_, D_);
    gemm_wmma<false, true, false, false><<<gg(BT_, HK_), blk, 0, stream>>>(x, lWk, lbk, lK, BT_, HK_, D_);
    gemm_wmma<false, true, false, false><<<gg(BT_, HK_), blk, 0, stream>>>(x, lWv, lbv, lV, BT_, HK_, D_);
    local_attn_kernel<<<dim3(1024), dim3(128), 0, stream>>>(lQ, lK, lV, lO);
    gemm_wmma<true, false, false, false><<<gg(BT_, D_), blk, 0, stream>>>(lO, lWo, lbo, attn, BT_, D_, HK_);

    // ---- Global branch: g2 = MHA(g, x) ----
    gemm_wmma<false, true, false, false><<<gg(BT_, HK_), blk, 0, stream>>>(x, gWk, gbk, gK, BT_, HK_, D_);
    gemm_wmma<false, true, false, false><<<gg(BT_, HK_), blk, 0, stream>>>(x, gWv, gbv, gV, BT_, HK_, D_);
    gather_g_kernel<<<dim3(16), dim3(256), 0, stream>>>(x, gbuf);
    gemm_wmma<false, false, false, false><<<gg(16, HK_), blk, 0, stream>>>(gbuf, gWq, gbq, gQ, 16, HK_, D_);
    g2_attn_kernel<<<dim3(128), dim3(256), 0, stream>>>(gQ, gK, gV, oG);
    gemm_wmma<true, false, false, false><<<gg(16, D_), blk, 0, stream>>>(oG, gWo, gbo, g2, 16, D_, HK_);

    // ---- Global branch: gx = MHA(x, g2), accumulated into attn ----
    gemm_wmma<false, false, false, false><<<gg(16, HK_), blk, 0, stream>>>(g2, gWk, gbk, K2, 16, HK_, D_);
    gemm_wmma<false, false, false, false><<<gg(16, HK_), blk, 0, stream>>>(g2, gWv, gbv, V2, 16, HK_, D_);
    gemm_wmma<false, true, false, false><<<gg(BT_, HK_), blk, 0, stream>>>(x, gWq, gbq, Q2, BT_, HK_, D_);
    gx_attn_kernel<<<dim3(BT_ * H_ / 8), dim3(256), 0, stream>>>(Q2, K2, V2, o2);
    gemm_wmma<true, false, false, true><<<gg(BT_, D_), blk, 0, stream>>>(o2, gWo, gbo, attn, BT_, D_, HK_);

    // ---- LN1 + FFN + LN2 ----
    ln_add_kernel<<<dim3(BT_ / 8), dim3(256), 0, stream>>>(x, attn, ln1g, ln1b, out1, BT_);
    gemm_wmma<false, true, true, false><<<gg(BT_, FF_), blk, 0, stream>>>(out1, fW1, fb1, h1, BT_, FF_, D_);
    gemm_wmma<true, false, false, false><<<gg(BT_, D_), blk, 0, stream>>>(h1, fW2, fb2, ffn, BT_, D_, FF_);
    ln_add_kernel<<<dim3(BT_ / 8), dim3(256), 0, stream>>>(out1, ffn, ln2g, ln2b, (float*)d_out, BT_);
}
